// TokenChoiceTopKRouter_10385230922011
// MI455X (gfx1250) — matrix operand order repaired and checked by reading, not compiled, not run on hardware
//
#include <hip/hip_runtime.h>
#include <hip/hip_bf16.h>

typedef __attribute__((ext_vector_type(16))) __bf16 bf16x16;
typedef __attribute__((ext_vector_type(8)))  __bf16 bf16x8;
typedef __attribute__((ext_vector_type(4)))  __bf16 bf16x4;
typedef __attribute__((ext_vector_type(8)))  float  f32x8;
typedef __attribute__((ext_vector_type(4)))  float  f32x4;

#define T_TOKENS 16384
#define DIMK     4096
#define NEXP     256
#define TOPK     8

// ---------------------------------------------------------------------------
// Kernel 0: split gate weights fp32 -> bf16 hi/lo (w = hi + lo to ~bf16^2 prec)
// ---------------------------------------------------------------------------
__global__ __launch_bounds__(256) void k_split_w(const float* __restrict__ w,
                                                 __bf16* __restrict__ whi,
                                                 __bf16* __restrict__ wlo) {
  int i = (blockIdx.x * 256 + threadIdx.x) * 4;
  f32x4 v = *(const f32x4*)(w + i);
  bf16x4 h, l;
#pragma unroll
  for (int j = 0; j < 4; j++) {
    __bf16 hb = (__bf16)v[j];
    h[j] = hb;
    l[j] = (__bf16)(v[j] - (float)hb);
  }
  *(bf16x4*)(whi + i) = h;
  *(bf16x4*)(wlo + i) = l;
}

__global__ void k_zero_counts(int* p, int n) {
  int i = blockIdx.x * 256 + threadIdx.x;
  if (i < n) p[i] = 0;
}

// ---------------------------------------------------------------------------
// Kernel 1: logits GEMM  (T x D) * (D x E)  via bf16 hi/lo split WMMA.
// Workgroup = 256 threads (8 waves). Tile: M=64 tokens, N=128 experts.
// Wave w: N-tile = w (16 experts), loops over 4 M-tiles.
// x is converted fp32 -> bf16(hi,lo) cooperatively into LDS (double buffered)
// already laid out as A fragments in contiguous k order (lane half h holds k 16h..16h+15).
// ---------------------------------------------------------------------------
__global__ __launch_bounds__(256) void k_gemm_logits(
    const float* __restrict__ x, const __bf16* __restrict__ whi,
    const __bf16* __restrict__ wlo, float* __restrict__ logits) {
  // [buf][hi/lo][m-tile][lane][elem]
  __shared__ __attribute__((aligned(64))) __bf16 lds[2][2][4][32][16];

  const int tid   = threadIdx.x;
  const int lane  = tid & 31;
  const int wid   = tid >> 5;
  const int mBase = blockIdx.x * 64;
  const int n0    = blockIdx.y * 128 + wid * 16;

  // --- cooperative A-staging decomposition: 256 threads cover 64 rows x 4
  //     8-wide K-chunks. Chunk j maps to (lane,elem-half), matching B's K order:
  //     lanes0-15: K0-7 -> elems0-7, K8-15 -> elems8-15
  //     lanes16-31: K16-23 -> elems0-7, K24-31 -> elems8-15
  const int r     = tid >> 2;                    // row 0..63
  const int j     = tid & 3;                     // K-chunk
  const int sm    = r >> 4;                      // m-tile
  const int slane = (r & 15) + ((j >> 1) << 4);  // dest lane slot
  const int soff  = (j & 1) << 3;                // elem half (0 or 8)
  const float* xrow = x + (size_t)(mBase + r) * DIMK + j * 8;

  // --- B fragment pointers (B 32x16, this code's contiguous k order: lanes0-15 K0-15, lanes16-31 K16-31)
  const int col = lane & 15;
  const int kb  = (lane >> 4) << 4;  // 0 or 16
  const __bf16* bhip = whi + (size_t)(n0 + col) * DIMK + kb;
  const __bf16* blop = wlo + (size_t)(n0 + col) * DIMK + kb;

  f32x8 acc[4] = {};

  // prologue: stage K-step 0 into buffer 0
  {
    f32x4 v0 = *(const f32x4*)(xrow + 0);
    f32x4 v1 = *(const f32x4*)(xrow + 4);
    bf16x8 h, l;
#pragma unroll
    for (int e = 0; e < 4; e++) {
      __bf16 hb = (__bf16)v0[e]; h[e]     = hb; l[e]     = (__bf16)(v0[e] - (float)hb);
      __bf16 h2 = (__bf16)v1[e]; h[4 + e] = h2; l[4 + e] = (__bf16)(v1[e] - (float)h2);
    }
    *(bf16x8*)&lds[0][0][sm][slane][soff] = h;
    *(bf16x8*)&lds[0][1][sm][slane][soff] = l;
  }

  int buf = 0;
#pragma unroll 1
  for (int ks = 0; ks < DIMK / 32; ++ks) {
    __syncthreads();
    const int k0 = ks * 32;

    // prefetch next x chunk to registers while computing current step
    f32x4 v0 = {}, v1 = {};
    if (ks + 1 < DIMK / 32) {
      v0 = *(const f32x4*)(xrow + k0 + 32);
      v1 = *(const f32x4*)(xrow + k0 + 36);
    }

    // B fragments (hi/lo) for this wave's expert tile (L2-resident)
    bf16x16 bh = *(const bf16x16*)(bhip + k0);
    bf16x16 bl = *(const bf16x16*)(blop + k0);

#pragma unroll
    for (int m = 0; m < 4; m++) {
      bf16x16 ah = *(const bf16x16*)&lds[buf][0][m][lane][0];
      bf16x16 al = *(const bf16x16*)&lds[buf][1][m][lane][0];
      acc[m] = __builtin_amdgcn_wmma_f32_16x16x32_bf16(false, ah, false, bh,
                                                       (short)0, acc[m], false, false);
      acc[m] = __builtin_amdgcn_wmma_f32_16x16x32_bf16(false, ah, false, bl,
                                                       (short)0, acc[m], false, false);
      acc[m] = __builtin_amdgcn_wmma_f32_16x16x32_bf16(false, al, false, bh,
                                                       (short)0, acc[m], false, false);
    }

    // convert + store next chunk into the other buffer
    if (ks + 1 < DIMK / 32) {
      bf16x8 h, l;
#pragma unroll
      for (int e = 0; e < 4; e++) {
        __bf16 hb = (__bf16)v0[e]; h[e]     = hb; l[e]     = (__bf16)(v0[e] - (float)hb);
        __bf16 h2 = (__bf16)v1[e]; h[4 + e] = h2; l[4 + e] = (__bf16)(v1[e] - (float)h2);
      }
      *(bf16x8*)&lds[buf ^ 1][0][sm][slane][soff] = h;
      *(bf16x8*)&lds[buf ^ 1][1][sm][slane][soff] = l;
    }
    buf ^= 1;
  }

  // C/D 32-bit 16x16 layout: lane = N, VGPR v = M row v (+8 for lanes 16-31)
  const int rowOff = (lane >> 4) << 3;
#pragma unroll
  for (int m = 0; m < 4; m++) {
#pragma unroll
    for (int v = 0; v < 8; v++) {
      logits[(size_t)(mBase + 16 * m + v + rowOff) * NEXP + n0 + col] = acc[m][v];
    }
  }
}

// ---------------------------------------------------------------------------
// Kernel 2: node-limited top-k routing. One wave32 per token.
// Lane l owns experts 8l..8l+7 (all inside group l/4).
// ---------------------------------------------------------------------------
__global__ __launch_bounds__(256) void k_route(const float* __restrict__ logits,
                                               const float* __restrict__ bias,
                                               float* __restrict__ outS,
                                               int* __restrict__ outI,
                                               int* __restrict__ counts) {
  const int lane = threadIdx.x & 31;
  const int t    = blockIdx.x * 8 + (threadIdx.x >> 5);
  const float NEG = -__builtin_inff();

  const float* lrow = logits + (size_t)t * NEXP + lane * 8;
  f32x4 l0 = *(const f32x4*)(lrow);
  f32x4 l1 = *(const f32x4*)(lrow + 4);
  f32x4 b0 = *(const f32x4*)(bias + lane * 8);
  f32x4 b1 = *(const f32x4*)(bias + lane * 8 + 4);

  float s[8], b[8], mb[8];
#pragma unroll
  for (int q = 0; q < 4; q++) {
    s[q]     = 1.0f / (1.0f + expf(-l0[q]));
    s[4 + q] = 1.0f / (1.0f + expf(-l1[q]));
    b[q]     = s[q] + b0[q];
    b[4 + q] = s[4 + q] + b1[q];
  }

  // local top-2 of biased scores
  float m1 = NEG, m2 = NEG;
#pragma unroll
  for (int q = 0; q < 8; q++) {
    float v = b[q];
    if (v > m1) { m2 = m1; m1 = v; } else if (v > m2) { m2 = v; }
  }
  // merge top-2 pairs across the 4 lanes of each group
#pragma unroll
  for (int d = 1; d <= 2; d <<= 1) {
    float o1 = __shfl_xor(m1, d, 32);
    float o2 = __shfl_xor(m2, d, 32);
    float n1 = fmaxf(m1, o1);
    float n2 = fmaxf(fminf(m1, o1), fmaxf(m2, o2));
    m1 = n1; m2 = n2;
  }
  float gscore = m1 + m2;
  const int g = lane >> 2;

  // rank my group among the 8 group scores (tie -> lower group index kept)
  int rank = 0;
#pragma unroll
  for (int gg = 0; gg < 8; gg++) {
    float og = __shfl(gscore, gg * 4, 32);
    rank += (og > gscore) || (og == gscore && gg < g);
  }
  const bool keep = rank < 4;
#pragma unroll
  for (int q = 0; q < 8; q++) mb[q] = keep ? b[q] : NEG;

  // iterative top-8 (tie -> lowest expert index, matching jax top_k)
  float ws[8]; int ids[8]; float denom = 0.0f;
#pragma unroll
  for (int rsel = 0; rsel < 8; rsel++) {
    float v = NEG; int idx = 0x7FFFFFFF;
#pragma unroll
    for (int q = 0; q < 8; q++) {
      if (mb[q] > v) { v = mb[q]; idx = lane * 8 + q; }
    }
#pragma unroll
    for (int d = 16; d >= 1; d >>= 1) {
      float ov = __shfl_xor(v, d, 32);
      int   oi = __shfl_xor(idx, d, 32);
      if (ov > v || (ov == v && oi < idx)) { v = ov; idx = oi; }
    }
    const int owner = idx >> 3, slot = idx & 7;  // wave-uniform
    float sv = 0.0f;
#pragma unroll
    for (int q = 0; q < 8; q++) if (slot == q) sv = s[q];  // original score
    float wsc = __shfl(sv, owner, 32);
    ws[rsel] = wsc; ids[rsel] = idx; denom += wsc;
#pragma unroll
    for (int q = 0; q < 8; q++)
      if (lane == owner && slot == q) mb[q] = NEG;  // retire winner
  }

  const float scale = 2.5f / (denom + 1e-20f);
#pragma unroll
  for (int rsel = 0; rsel < 8; rsel++) {
    if (lane == rsel) {
      outS[(size_t)t * TOPK + rsel] = ws[rsel] * scale;
      outI[(size_t)t * TOPK + rsel] = ids[rsel];
      atomicAdd(counts + ids[rsel], 1);
    }
  }
}

// ---------------------------------------------------------------------------
extern "C" void kernel_launch(void* const* d_in, const int* in_sizes, int n_in,
                              void* d_out, int out_size, void* d_ws, size_t ws_size,
                              hipStream_t stream) {
  const float* x  = (const float*)d_in[0];       // (T, D)
  const float* gw = (const float*)d_in[1];       // (E, D)
  const float* eb = (const float*)d_in[2];       // (E,)

  char* ws = (char*)d_ws;
  __bf16* whi   = (__bf16*)ws;                            // 2 MB
  __bf16* wlo   = (__bf16*)(ws + (size_t)2 * 1024 * 1024); // 2 MB
  float*  logit = (float*)(ws + (size_t)4 * 1024 * 1024);  // 16 MB

  float* outS   = (float*)d_out;                          // (T, 8) f32
  int*   outI   = (int*)d_out + (size_t)T_TOKENS * TOPK;  // (T, 8) i32
  int*   counts = (int*)d_out + (size_t)2 * T_TOKENS * TOPK;  // (E,) i32

  k_split_w<<<(NEXP * DIMK) / 1024, 256, 0, stream>>>(gw, whi, wlo);
  k_zero_counts<<<1, 256, 0, stream>>>(counts, NEXP);

  dim3 gGemm(T_TOKENS / 64, NEXP / 128);
  k_gemm_logits<<<gGemm, 256, 0, stream>>>(x, whi, wlo, logit);

  k_route<<<T_TOKENS / 8, 256, 0, stream>>>(logit, eb, outS, outI, counts);
}
